// DPLSTM_model_72533407695218
// MI455X (gfx1250) — matrix operand order repaired and checked by reading, not compiled, not run on hardware
//
#include <hip/hip_runtime.h>
#include <hip/hip_bf16.h>

typedef __bf16 bf16;
typedef __attribute__((ext_vector_type(16))) __bf16 v16bf;
typedef __attribute__((ext_vector_type(8)))  float  v8f;

#define B_      64
#define T_      1024
#define EMB_    512
#define HID_    512
#define NWG     32
#define THREADS 256

// ---------------- workspace layout (bytes) ----------------
static const size_t OFF_WIH   = 0;                 // 2048*512 bf16 = 2 MB
static const size_t OFF_WHH   = 2097152;           // 2 MB
static const size_t OFF_BIAS  = 4194304;           // 2048 f32
static const size_t OFF_H     = 4202496;           // 64*512 bf16 = 64 KB
static const size_t OFF_HLAST = 4268032;           // 64*512 f32 = 128 KB
static const size_t OFF_BAR   = 4399104;           // barrier ints
static const size_t OFF_E     = 4400128;           // [T][B][EMB] bf16 = 64 MB

// ---------------- WMMA fragment loaders (ISA 7.12.2 layouts) ----------------
union FragU { v16bf v; float4 f4[2]; };

// A 16x32 bf16 tile from row-major [rows][ldk]:
// lane l(0..15) holds row row0+l; hi=lane>>4.
// elems 0..7  = K[kbase +      hi*8 .. +7]
// elems 8..15 = K[kbase + 16 + hi*8 .. +7]
__device__ inline v16bf load_a_frag(const bf16* __restrict__ base, int row0, int kbase, int ldk) {
  int lane = threadIdx.x & 31;
  int l = lane & 15, hi = lane >> 4;
  const bf16* p = base + (size_t)(row0 + l) * ldk;
  FragU u;
  u.f4[0] = *(const float4*)(p + kbase + hi * 8);
  u.f4[1] = *(const float4*)(p + kbase + 16 + hi * 8);
  return u.v;
}

// B 32x16 bf16 tile = W^T slice, W row-major [2048][ldk]:
// lane l(0..15) holds column n=ncol0+l (a row of W); elems 0..7 = K[kbase+hi*8 .. +7], elems 8..15 = K[kbase+16+hi*8 .. +7].
__device__ inline v16bf load_b_frag(const bf16* __restrict__ w, int ncol0, int kbase, int ldk) {
  int lane = threadIdx.x & 31;
  int l = lane & 15, hi = lane >> 4;
  const bf16* p = w + (size_t)(ncol0 + l) * ldk + kbase + hi * 8;
  FragU u;
  u.f4[0] = *(const float4*)(p);
  u.f4[1] = *(const float4*)(p + 16);
  return u.v;
}

__device__ inline float sigmoidf_(float x) { return 1.0f / (1.0f + __expf(-x)); }
__device__ inline float tanhf_(float x) {
  x = fminf(fmaxf(x, -15.0f), 15.0f);
  float e2 = __expf(-2.0f * x);
  return (1.0f - e2) / (1.0f + e2);
}

// ---------------- prep: cast weights to bf16, combined bias, zero h & barrier ----------------
__global__ void prep_kernel(const float* __restrict__ w_ih, const float* __restrict__ w_hh,
                            const float* __restrict__ b_ih, const float* __restrict__ b_hh,
                            bf16* __restrict__ wih_b, bf16* __restrict__ whh_b,
                            float* __restrict__ bias, bf16* __restrict__ h_g,
                            int* __restrict__ bar) {
  const long NW = 2048L * 512L;
  long stride = (long)gridDim.x * blockDim.x;
  for (long i = (long)blockIdx.x * blockDim.x + threadIdx.x; i < NW; i += stride) {
    wih_b[i] = (bf16)w_ih[i];
    whh_b[i] = (bf16)w_hh[i];
    if (i < 2048) bias[i] = b_ih[i] + b_hh[i];
    if (i < B_ * HID_) h_g[i] = (bf16)0.0f;
    if (i < 8) bar[i] = 0;
  }
}

// ---------------- embedding gather: e[t][b][:] = bf16(emb[x[b][t]][:]) ----------------
__global__ void embed_kernel(const int* __restrict__ x, const float* __restrict__ emb,
                             bf16* __restrict__ e) {
  const long N = (long)B_ * T_ * EMB_;
  long stride = (long)gridDim.x * blockDim.x;
  for (long i = (long)blockIdx.x * blockDim.x + threadIdx.x; i < N; i += stride) {
    long b = i >> 19;            // / (T_*EMB_)
    long r = i & ((1L << 19) - 1);
    long t = r >> 9;             // / EMB_
    long j = r & 511;
    int idx = x[b * T_ + t];
    float v = emb[(long)idx * EMB_ + j];
    e[((t * B_ + b) << 9) + j] = (bf16)v;
  }
}

// ---------------- persistent cooperative LSTM scan ----------------
__global__ __launch_bounds__(THREADS, 1)
void lstm_scan_kernel(const bf16* __restrict__ e,     // [T][B][EMB]
                      const bf16* __restrict__ wih,   // [2048][512] bf16
                      const bf16* __restrict__ whh,   // [2048][512] bf16
                      const float* __restrict__ bias, // [2048] = b_ih + b_hh
                      bf16* __restrict__ h_g,         // [B][HID] bf16, globally shared
                      float* __restrict__ h_last,     // [B][HID] f32
                      int* __restrict__ bar) {
  __shared__ float gates_s[B_][68]; // local cols: [i 0..15][f 16..31][g 32..47][o 48..63]
  __shared__ float c_s[B_][20];     // cell state for our 16 hidden cols

  const int wg   = blockIdx.x;      // hidden slice [wg*16, wg*16+16)
  const int tid  = threadIdx.x;
  const int wave = tid >> 5;
  const int lane = tid & 31;
  const int l    = lane & 15;
  const int hi   = lane >> 4;
  const int gate = wave & 3;        // 0:i 1:f 2:g 3:o
  const int m0   = (wave >> 2) * 32;          // rows m0..m0+31 (two 16-row tiles)
  const int col0 = gate * 512 + wg * 16;      // global gate-column base (N tile)

  for (int i = tid; i < B_ * 16; i += THREADS) c_s[i >> 4][i & 15] = 0.0f;
  __syncthreads();

  const float bv = bias[col0 + l];

  for (int t = 0; t < T_; ++t) {
    v8f acc0, acc1;
#pragma unroll
    for (int i = 0; i < 8; ++i) { acc0[i] = bv; acc1[i] = bv; }

    const bf16* et = e + (size_t)t * (B_ * EMB_);
    if (t + 1 < T_) { // prefetch next timestep's 64KB input block
      const bf16* en = et + B_ * EMB_;
      __builtin_prefetch(en + (size_t)tid * 128, 0, 0);
    }

    // gates += e_t @ wih^T  (K = EMB)
#pragma unroll 4
    for (int kc = 0; kc < EMB_; kc += 32) {
      v16bf bf = load_b_frag(wih, col0, kc, EMB_);
      v16bf a0 = load_a_frag(et, m0, kc, EMB_);
      v16bf a1 = load_a_frag(et, m0 + 16, kc, EMB_);
      acc0 = __builtin_amdgcn_wmma_f32_16x16x32_bf16(false, a0, false, bf, (short)0, acc0, false, false);
      acc1 = __builtin_amdgcn_wmma_f32_16x16x32_bf16(false, a1, false, bf, (short)0, acc1, false, false);
    }
    // gates += h @ whh^T  (K = HID)
#pragma unroll 4
    for (int kc = 0; kc < HID_; kc += 32) {
      v16bf bf = load_b_frag(whh, col0, kc, HID_);
      v16bf a0 = load_a_frag(h_g, m0, kc, HID_);
      v16bf a1 = load_a_frag(h_g, m0 + 16, kc, HID_);
      acc0 = __builtin_amdgcn_wmma_f32_16x16x32_bf16(false, a0, false, bf, (short)0, acc0, false, false);
      acc1 = __builtin_amdgcn_wmma_f32_16x16x32_bf16(false, a1, false, bf, (short)0, acc1, false, false);
    }

    // stage D tiles to LDS: elem r -> row = base + r + 8*hi, col = lane%16
#pragma unroll
    for (int r = 0; r < 8; ++r) {
      gates_s[m0 + r + 8 * hi][gate * 16 + l]      = acc0[r];
      gates_s[m0 + 16 + r + 8 * hi][gate * 16 + l] = acc1[r];
    }
    __syncthreads();

    // elementwise cell/hidden update for our 16 hidden columns
    for (int i = tid; i < B_ * 16; i += THREADS) {
      int b = i >> 4, q = i & 15;
      float is = sigmoidf_(gates_s[b][q]);
      float fs = sigmoidf_(gates_s[b][16 + q]);
      float gt = tanhf_(gates_s[b][32 + q]);
      float os = sigmoidf_(gates_s[b][48 + q]);
      float c  = fs * c_s[b][q] + is * gt;
      c_s[b][q] = c;
      float h = os * tanhf_(c);
      h_g[(size_t)b * HID_ + wg * 16 + q] = (bf16)h;
      if (t == T_ - 1) h_last[(size_t)b * HID_ + wg * 16 + q] = h;
    }

    // grid-wide generation barrier so every WG sees the full new h
    __threadfence();
    __syncthreads();
    if (tid == 0) {
      volatile int* gen = bar + 1;
      int g = *gen;
      if (atomicAdd(bar, 1) == NWG - 1) {
        bar[0] = 0;
        __threadfence();
        atomicExch((int*)(bar + 1), g + 1);
      } else {
        while (*gen == g) __builtin_amdgcn_s_sleep(1);
      }
    }
    __syncthreads();
    __threadfence();
  }
}

// ---------------- output head: out[64,2] = h_last @ w_out^T + b_out ----------------
__global__ void out_kernel(const float* __restrict__ h_last, const float* __restrict__ w_out,
                           const float* __restrict__ b_out, float* __restrict__ out) {
  int tid = threadIdx.x;
  if (tid < 128) {
    int b = tid >> 1, o = tid & 1;
    float s = b_out[o];
    const float* hp = h_last + b * HID_;
    const float* wp = w_out + o * HID_;
#pragma unroll 8
    for (int k = 0; k < HID_; ++k) s += hp[k] * wp[k];
    out[b * 2 + o] = s;
  }
}

extern "C" void kernel_launch(void* const* d_in, const int* in_sizes, int n_in,
                              void* d_out, int out_size, void* d_ws, size_t ws_size,
                              hipStream_t stream) {
  (void)in_sizes; (void)n_in; (void)out_size; (void)ws_size;
  const int*   x     = (const int*)d_in[0];
  const float* emb   = (const float*)d_in[1];
  const float* w_ih  = (const float*)d_in[2];
  const float* w_hh  = (const float*)d_in[3];
  const float* b_ih  = (const float*)d_in[4];
  const float* b_hh  = (const float*)d_in[5];
  const float* w_out = (const float*)d_in[6];
  const float* b_out = (const float*)d_in[7];
  float* out = (float*)d_out;

  char* ws = (char*)d_ws;
  bf16*  wih_b  = (bf16*)(ws + OFF_WIH);
  bf16*  whh_b  = (bf16*)(ws + OFF_WHH);
  float* bias   = (float*)(ws + OFF_BIAS);
  bf16*  h_g    = (bf16*)(ws + OFF_H);
  float* h_last = (float*)(ws + OFF_HLAST);
  int*   bar    = (int*)(ws + OFF_BAR);
  bf16*  e      = (bf16*)(ws + OFF_E);

  prep_kernel<<<2048, 256, 0, stream>>>(w_ih, w_hh, b_ih, b_hh, wih_b, whh_b, bias, h_g, bar);
  embed_kernel<<<8192, 256, 0, stream>>>(x, emb, e);
  lstm_scan_kernel<<<NWG, THREADS, 0, stream>>>(e, wih_b, whh_b, bias, h_g, h_last, bar);
  out_kernel<<<1, 128, 0, stream>>>(h_last, w_out, b_out, out);
}
